// EncoderRecurrent_35914516529760
// MI455X (gfx1250) — matrix operand order repaired and checked by reading, not compiled, not run on hardware
//
#include <hip/hip_runtime.h>

// ---------------------------------------------------------------------------
// CDNA5 (gfx1250) WMMA implementation of the sparse recurrent encoder.
// Activations kept in f16 between layers (L2-resident: 8MB @ N0); all matmuls
// use v_wmma_f32_16x16x32_f16 with f32 accumulation.  Sparse-conv weights are
// staged into LDS by the Tensor Data Mover (tensor_load_to_lds, TENSORcnt),
// double-buffered so DMA overlaps the WMMA chain; B fragments come from LDS.
// ---------------------------------------------------------------------------

typedef __attribute__((ext_vector_type(16))) _Float16 v16h;
typedef __attribute__((ext_vector_type(8)))  _Float16 v8h;
typedef __attribute__((ext_vector_type(8)))  float    v8f;
typedef __attribute__((ext_vector_type(4))) unsigned int u32x4;
typedef __attribute__((ext_vector_type(8))) int          i32x8;
typedef __attribute__((ext_vector_type(4))) int          i32x4;

#define WMMA_F16(a, b, c) \
  __builtin_amdgcn_wmma_f32_16x16x32_f16(false, (a), false, (b), (short)0, (c), false, false)

#if defined(__AMDGCN__) && __has_builtin(__builtin_amdgcn_tensor_load_to_lds)
#define SPCONV_TDM 1
#else
#define SPCONV_TDM 0
#endif

static __device__ __forceinline__ v16h cat8(v8h lo, v8h hi) {
  return __builtin_shufflevector(lo, hi, 0, 1, 2, 3, 4, 5, 6, 7,
                                         8, 9, 10, 11, 12, 13, 14, 15);
}

// A fragment (16x32 f16, ISA layout): row = lane&15, lane-group lg = lane>>4.
// Elements 0..7  hold K = ks*32 + lg*8 + e
// Elements 8..15 hold K = ks*32 + 16 + lg*8 + (e-8)
static __device__ __forceinline__ v16h frag_a(const _Float16* __restrict__ rowp,
                                              int ks, int lg) {
  const _Float16* p = rowp + ks * 32 + lg * 8;
  v8h lo = *(const v8h*)(p);
  v8h hi = *(const v8h*)(p + 16);
  return cat8(lo, hi);
}

// B fragment (32x16 f16, ISA layout): column = lane&15, K = lg*8 + e (e<8),
// 16 + lg*8 + (e-8) (e>=8) within the 32-wide K slice.  Weights pre-transposed
// to [n][k] (two b128 loads, global or LDS).
static __device__ __forceinline__ v16h frag_b(const _Float16* p) {
  v8h lo = *(const v8h*)(p);
  v8h hi = *(const v8h*)(p + 16);
  return cat8(lo, hi);
}

// A fragment from an f32 row (K = 32, single slice): used by the gate.
static __device__ __forceinline__ v16h frag_a_f32(const float* __restrict__ rowp,
                                                  int lg) {
  v16h a;
#pragma unroll
  for (int e = 0; e < 8; ++e) a[e] = (_Float16)rowp[lg * 8 + e];
#pragma unroll
  for (int e = 0; e < 8; ++e) a[e + 8] = (_Float16)rowp[16 + lg * 8 + e];
  return a;
}

#if SPCONV_TDM
// Issue a TDM load of a weight chunk: 2D tile, dim0 = 4096 halves (one tap,
// [d][c]), dim1 = rows taps, contiguous (stride 4096).  D# packed per CDNA5
// ISA ch.8: group0 = {count, lds_addr, global_addr, type=2},
// group1 = {mask=0, data_size=2B, dims/tile dims/stride}.
static __device__ __forceinline__ void tdm_load_weights(
    const _Float16* gsrc, unsigned lds_byte_off, int rows) {
  unsigned long long ga = (unsigned long long)(size_t)gsrc;
  u32x4 g0;
  g0[0] = 1u;                                    // count=1 (valid descriptor)
  g0[1] = lds_byte_off;                          // lds_addr (bytes)
  g0[2] = (unsigned)(ga & 0xffffffffull);        // global_addr[31:0]
  g0[3] = (unsigned)((ga >> 32) & 0x01ffffffull) // global_addr[56:32]
        | (2u << 30);                            // type = 2 ("image")
  unsigned ur = (unsigned)rows;
  i32x8 g1;
  g1[0] = (int)(1u << 16);                 // workgroup_mask=0, data_size=1 (2B)
  g1[1] = (int)((4096u & 0xffffu) << 16);  // tensor_dim0[15:0] = 4096
  g1[2] = (int)((ur & 0xffffu) << 16);     // tensor_dim0 hi=0 | tensor_dim1 lo
  g1[3] = (int)(4096u << 16);              // tensor_dim1 hi=0 | tile_dim0=4096
  g1[4] = (int)(ur & 0xffffu);             // tile_dim1 = rows, tile_dim2 = 0
  g1[5] = 4096;                            // tensor_dim0_stride[31:0]
  g1[6] = 0;                               // stride0 hi | tensor_dim1_stride lo
  g1[7] = 0;
  i32x4 gz = {0, 0, 0, 0};
#if defined(__clang_major__) && (__clang_major__ >= 23)
  i32x8 gz8 = {0, 0, 0, 0, 0, 0, 0, 0};
  __builtin_amdgcn_tensor_load_to_lds(g0, g1, gz, gz, gz8, 0);
#else
  __builtin_amdgcn_tensor_load_to_lds(g0, g1, gz, gz, 0);
#endif
}
#endif  // SPCONV_TDM

// ---------------------------------------------------------------------------
// Gather-GEMM sparse conv:  fout[n,:] = relu( sum_k fin[nmap[n,k],:] @ W[k] + b )
// Block = 8 waves x 2 row-tiles (256 rows).  Weights staged into LDS in
// 9-tap chunks (72KB x 2 buffers of the WGP's 320KB) by the TDM; each B
// fragment (ds_load_b128 x2) feeds one WMMA per row tile.
// ---------------------------------------------------------------------------
constexpr int TPW    = 2;   // row tiles per wave
constexpr int KCHUNK = 9;   // taps staged per chunk (27 = 3*9; 8 = 1 chunk)

__global__ void __launch_bounds__(256)
spconv_wmma(const _Float16* __restrict__ fin, const int* __restrict__ nmap,
            int Kt, const _Float16* __restrict__ Wt,
            const float* __restrict__ bias, _Float16* __restrict__ fout,
            int Nout) {
  __shared__ _Float16 wlds[2][KCHUNK * 64 * 64];  // 2 x 73,728 bytes

  int tid  = threadIdx.x;
  int lane = tid & 31;
  int wv   = tid >> 5;  // 0..7
  int m = lane & 15, lg = lane >> 4;

  int tile0 = blockIdx.x * 16 + wv * TPW;
  int n0[TPW];
  const int* nm[TPW];
  int idx[TPW];
  v8f acc[TPW][4];
#pragma unroll
  for (int t = 0; t < TPW; ++t) {
    n0[t] = (tile0 + t) << 4;
    nm[t] = nmap + (size_t)(n0[t] + m) * Kt;
    idx[t] = nm[t][0];
#pragma unroll
    for (int nt = 0; nt < 4; ++nt) acc[t][nt] = (v8f){};
  }

  int nchunks = (Kt + KCHUNK - 1) / KCHUNK;

#if SPCONV_TDM
  if (wv == 0) {  // TDM ignores EXEC; scalar branch -> single DMA issue
    int kc0 = (Kt < KCHUNK) ? Kt : KCHUNK;
    tdm_load_weights(Wt, (unsigned)(size_t)(void*)&wlds[0][0], kc0);
  }
#endif

  for (int ci = 0; ci < nchunks; ++ci) {
    int k0 = ci * KCHUNK;
    int kc = (Kt - k0 < KCHUNK) ? (Kt - k0) : KCHUNK;
    const _Float16* buf = &wlds[ci & 1][0];

#if SPCONV_TDM
    if (wv == 0) {
      if (ci + 1 < nchunks) {
        // Prefetch next chunk into the other buffer, then wait for current
        // (TDM ops from one wave complete in order -> tensorcnt<=1 is enough).
        int k1 = k0 + KCHUNK;
        int kc1 = (Kt - k1 < KCHUNK) ? (Kt - k1) : KCHUNK;
        tdm_load_weights(Wt + ((size_t)k1 << 12),
                         (unsigned)(size_t)(void*)&wlds[(ci + 1) & 1][0], kc1);
        __builtin_amdgcn_s_wait_tensorcnt(1);
      } else {
        __builtin_amdgcn_s_wait_tensorcnt(0);
      }
    }
#else
    {  // cooperative fallback stage (also used by the host compile pass)
      const v8h* src = (const v8h*)(Wt + ((size_t)k0 << 12));
      v8h* dst = (v8h*)buf;
      int nvec = kc << 9;  // kc * 4096 halves / 8
      for (int i = tid; i < nvec; i += 256) dst[i] = src[i];
    }
#endif
    __syncthreads();

    for (int kk = 0; kk < kc; ++kk) {
      int k = k0 + kk;
      const _Float16* rowp[TPW];
#pragma unroll
      for (int t = 0; t < TPW; ++t) rowp[t] = fin + (size_t)idx[t] * 64;
      if (k + 1 < Kt) {
#pragma unroll
        for (int t = 0; t < TPW; ++t) {
          idx[t] = nm[t][k + 1];
          __builtin_prefetch(fin + (size_t)idx[t] * 64, 0, 0);  // global_prefetch_b8
        }
      }
      const _Float16* wk = buf + (kk << 12);
#pragma unroll
      for (int ks = 0; ks < 2; ++ks) {
        v16h a[TPW];
#pragma unroll
        for (int t = 0; t < TPW; ++t) a[t] = frag_a(rowp[t], ks, lg);
        const _Float16* wb = wk + ks * 32 + lg * 8;
#pragma unroll
        for (int nt = 0; nt < 4; ++nt) {
          v16h b = frag_b(wb + (nt * 16 + m) * 64);  // ds_load_b128 x2
#pragma unroll
          for (int t = 0; t < TPW; ++t) acc[t][nt] = WMMA_F16(a[t], b, acc[t][nt]);
        }
      }
    }
    __syncthreads();
  }

  float b0 = bias[m], b1 = bias[16 + m], b2 = bias[32 + m], b3 = bias[48 + m];
#pragma unroll
  for (int t = 0; t < TPW; ++t) {
#pragma unroll
    for (int r = 0; r < 8; ++r) {
      int orow = n0[t] + lg * 8 + r;  // C/D layout: VGPR r -> row r (+8 lanes>=16)
      _Float16* op = fout + (size_t)orow * 64 + m;
      op[0]  = (_Float16)fmaxf(acc[t][0][r] + b0, 0.0f);
      op[16] = (_Float16)fmaxf(acc[t][1][r] + b1, 0.0f);
      op[32] = (_Float16)fmaxf(acc[t][2][r] + b2, 0.0f);
      op[48] = (_Float16)fmaxf(acc[t][3][r] + b3, 0.0f);
    }
  }
}

// ---------------------------------------------------------------------------
// hx = cx @ Wout + bout   ([N,64] f16 @ [64,32]) -> f32 output region
// Wout pre-transposed to [d=32][c=64] f16.
// ---------------------------------------------------------------------------
__global__ void __launch_bounds__(128)
gemm_out_wmma(const _Float16* __restrict__ fin, const _Float16* __restrict__ Wt,
              const float* __restrict__ bout, float* __restrict__ out, int N) {
  int wave = blockIdx.x * (blockDim.x >> 5) + (threadIdx.x >> 5);
  int lane = threadIdx.x & 31;
  int n0 = wave << 4;
  if (n0 >= N) return;
  int m = lane & 15, lg = lane >> 4;

  v8f acc0 = {}, acc1 = {};
  const _Float16* rowp = fin + (size_t)(n0 + m) * 64;
#pragma unroll
  for (int ks = 0; ks < 2; ++ks) {
    v16h a = frag_a(rowp, ks, lg);
    const _Float16* wb = Wt + ks * 32 + lg * 8;
    acc0 = WMMA_F16(a, frag_b(wb + (0 * 16 + m) * 64), acc0);
    acc1 = WMMA_F16(a, frag_b(wb + (1 * 16 + m) * 64), acc1);
  }

  float b0 = bout[m], b1 = bout[16 + m];
#pragma unroll
  for (int r = 0; r < 8; ++r) {
    int orow = n0 + lg * 8 + r;
    out[(size_t)orow * 32 + m]      = acc0[r] + b0;
    out[(size_t)orow * 32 + 16 + m] = acc1[r] + b1;
  }
}

// ---------------------------------------------------------------------------
// g = sigmoid(hx @ Wg + bg);  cxout = g * cxin
// hx: [N,32] f32, Wg pre-transposed to [h=64][o=32] f16.  K = 32 (one slice).
// ---------------------------------------------------------------------------
__global__ void __launch_bounds__(128)
gate_wmma(const float* __restrict__ hx, const _Float16* __restrict__ Wgt,
          const float* __restrict__ bg, const _Float16* __restrict__ cxin,
          _Float16* __restrict__ cxout, int N) {
  int wave = blockIdx.x * (blockDim.x >> 5) + (threadIdx.x >> 5);
  int lane = threadIdx.x & 31;
  int n0 = wave << 4;
  if (n0 >= N) return;
  int m = lane & 15, lg = lane >> 4;

  v16h a = frag_a_f32(hx + (size_t)(n0 + m) * 32, lg);
  const _Float16* wb = Wgt + lg * 8;
  v8f acc0 = {}, acc1 = {}, acc2 = {}, acc3 = {};
  acc0 = WMMA_F16(a, frag_b(wb + (0 * 16 + m) * 32), acc0);
  acc1 = WMMA_F16(a, frag_b(wb + (1 * 16 + m) * 32), acc1);
  acc2 = WMMA_F16(a, frag_b(wb + (2 * 16 + m) * 32), acc2);
  acc3 = WMMA_F16(a, frag_b(wb + (3 * 16 + m) * 32), acc3);

  float g0 = bg[m], g1 = bg[16 + m], g2 = bg[32 + m], g3 = bg[48 + m];
#pragma unroll
  for (int r = 0; r < 8; ++r) {
    int orow = n0 + lg * 8 + r;
    const _Float16* ci = cxin + (size_t)orow * 64 + m;
    _Float16* co = cxout + (size_t)orow * 64 + m;
    float t, g;
    t = acc0[r] + g0; g = 1.0f / (1.0f + __expf(-t)); co[0]  = (_Float16)(g * (float)ci[0]);
    t = acc1[r] + g1; g = 1.0f / (1.0f + __expf(-t)); co[16] = (_Float16)(g * (float)ci[16]);
    t = acc2[r] + g2; g = 1.0f / (1.0f + __expf(-t)); co[32] = (_Float16)(g * (float)ci[32]);
    t = acc3[r] + g3; g = 1.0f / (1.0f + __expf(-t)); co[48] = (_Float16)(g * (float)ci[48]);
  }
}

// ---------------------------------------------------------------------------
// Prep: weight transpose/convert  in[k][c][d] f32 -> out[k][d][c] f16
// ---------------------------------------------------------------------------
__global__ void __launch_bounds__(256)
transpose_kcd_f16(const float* __restrict__ in, _Float16* __restrict__ out,
                  int K, int C, int D) {
  int i = blockIdx.x * blockDim.x + threadIdx.x;
  int total = K * C * D;
  if (i >= total) return;
  int k = i / (C * D);
  int rem = i - k * (C * D);
  int c = rem / D;
  int d = rem - c * D;
  out[(size_t)k * C * D + (size_t)d * C + c] = (_Float16)in[i];
}

__global__ void __launch_bounds__(256)
cast_f32_to_f16(const float* __restrict__ in, _Float16* __restrict__ out, int n) {
  int i = blockIdx.x * blockDim.x + threadIdx.x;
  if (i < n) out[i] = (_Float16)in[i];
}

// ---------------------------------------------------------------------------
// Host orchestration
// ---------------------------------------------------------------------------
extern "C" void kernel_launch(void* const* d_in, const int* in_sizes, int n_in,
                              void* d_out, int out_size, void* d_ws,
                              size_t ws_size, hipStream_t stream) {
  constexpr int H = 64, OC = 32;
  constexpr int N0 = 65536, N1 = 16384, N2 = 4096, N3 = 1024;
  constexpr int K3 = 27, K2 = 8;

  const float* x    = (const float*)d_in[0];
  const float* Wg0  = (const float*)d_in[1];
  const float* bg0  = (const float*)d_in[2];
  const float* Wg1  = (const float*)d_in[3];
  const float* bg1  = (const float*)d_in[4];
  const float* Wout = (const float*)d_in[5];
  const float* bout = (const float*)d_in[6];
  const float* WA   = (const float*)d_in[7];
  const float* bA   = (const float*)d_in[8];
  const float* WB   = (const float*)d_in[9];
  const float* bB   = (const float*)d_in[10];
  const float* WD   = (const float*)d_in[11];
  const float* bD   = (const float*)d_in[12];
  const float* WC   = (const float*)d_in[13];
  const float* bC   = (const float*)d_in[14];
  const int* nmap0  = (const int*)d_in[15];
  const int* nmap1  = (const int*)d_in[16];
  const int* nmap2  = (const int*)d_in[17];
  const int* nmap3  = (const int*)d_in[18];
  const int* nd0    = (const int*)d_in[19];
  const int* nd1    = (const int*)d_in[20];
  const int* nd2    = (const int*)d_in[21];
  (void)in_sizes; (void)n_in; (void)out_size; (void)ws_size;

  // Workspace carve-up (halves).  Total ~17.5 MB.
  constexpr size_t SZ_W3 = (size_t)K3 * H * H;  // 110592
  constexpr size_t SZ_W2 = (size_t)K2 * H * H;  // 32768
  constexpr size_t SZ_WG = (size_t)H * OC;      // 2048
  constexpr size_t SZ_F  = (size_t)N0 * H;      // 4194304
  _Float16* ws   = (_Float16*)d_ws;
  _Float16* tWA  = ws;
  _Float16* tWB  = tWA + SZ_W3;
  _Float16* tWC  = tWB + SZ_W3;
  _Float16* tWD  = tWC + SZ_W3;
  _Float16* tWg0 = tWD + SZ_W2;
  _Float16* tWg1 = tWg0 + SZ_WG;
  _Float16* tWo  = tWg1 + SZ_WG;
  _Float16* fA   = tWo + SZ_WG;
  _Float16* fB   = fA + SZ_F;

  float* out = (float*)d_out;
  float* o0 = out;
  float* o1 = o0 + (size_t)N0 * OC;
  float* o2 = o1 + (size_t)N0 * OC;
  float* o3 = o2 + (size_t)N1 * OC;
  float* o4 = o3 + (size_t)N1 * OC;
  float* o5 = o4 + (size_t)N2 * OC;
  float* o6 = o5 + (size_t)N2 * OC;

  auto cdiv = [](int a, int b) { return (a + b - 1) / b; };

  // Weight prep (f32 -> f16, [k][c][d] -> [k][d][c])
  transpose_kcd_f16<<<cdiv(K3 * H * H, 256), 256, 0, stream>>>(WA, tWA, K3, H, H);
  transpose_kcd_f16<<<cdiv(K3 * H * H, 256), 256, 0, stream>>>(WB, tWB, K3, H, H);
  transpose_kcd_f16<<<cdiv(K3 * H * H, 256), 256, 0, stream>>>(WC, tWC, K3, H, H);
  transpose_kcd_f16<<<cdiv(K2 * H * H, 256), 256, 0, stream>>>(WD, tWD, K2, H, H);
  transpose_kcd_f16<<<cdiv(OC * H, 256), 256, 0, stream>>>(Wg0, tWg0, 1, OC, H);
  transpose_kcd_f16<<<cdiv(OC * H, 256), 256, 0, stream>>>(Wg1, tWg1, 1, OC, H);
  transpose_kcd_f16<<<cdiv(H * OC, 256), 256, 0, stream>>>(Wout, tWo, 1, H, OC);
  cast_f32_to_f16<<<cdiv(N0 * H, 256), 256, 0, stream>>>(x, fA, N0 * H);

  auto spconv = [&](const _Float16* fin, const int* nmap, int Kt,
                    const _Float16* Wt, const float* bias, _Float16* fout, int N) {
    spconv_wmma<<<N / 256, 256, 0, stream>>>(fin, nmap, Kt, Wt, bias, fout, N);
  };
  auto gemm_out = [&](const _Float16* fin, float* o, int N) {
    gemm_out_wmma<<<N / 64, 128, 0, stream>>>(fin, tWo, bout, o, N);
  };
  auto gate = [&](const float* hx, const _Float16* Wgt, const float* bg,
                  const _Float16* cin, _Float16* cout, int N) {
    gate_wmma<<<N / 64, 128, 0, stream>>>(hx, Wgt, bg, cin, cout, N);
  };

  gemm_out(fA, o0, N0);
  // scale 0
  gate(o0, tWg0, bg0, fA, fB, N0);
  spconv(fB, nmap0, K3, tWA, bA, fA, N0);
  spconv(fA, nmap0, K3, tWB, bB, fB, N0);
  gemm_out(fB, o1, N0);
  gate(o1, tWg1, bg1, fB, fA, N0);
  spconv(fA, nd0, K2, tWD, bD, fB, N1);
  spconv(fB, nmap1, K3, tWC, bC, fA, N1);
  gemm_out(fA, o2, N1);
  // scale 1
  gate(o2, tWg0, bg0, fA, fB, N1);
  spconv(fB, nmap1, K3, tWA, bA, fA, N1);
  spconv(fA, nmap1, K3, tWB, bB, fB, N1);
  gemm_out(fB, o3, N1);
  gate(o3, tWg1, bg1, fB, fA, N1);
  spconv(fA, nd1, K2, tWD, bD, fB, N2);
  spconv(fB, nmap2, K3, tWC, bC, fA, N2);
  gemm_out(fA, o4, N2);
  // scale 2
  gate(o4, tWg0, bg0, fA, fB, N2);
  spconv(fB, nmap2, K3, tWA, bA, fA, N2);
  spconv(fA, nmap2, K3, tWB, bB, fB, N2);
  gemm_out(fB, o5, N2);
  gate(o5, tWg1, bg1, fB, fA, N2);
  spconv(fA, nd2, K2, tWD, bD, fB, N3);
  spconv(fB, nmap3, K3, tWC, bC, fA, N3);
  gemm_out(fA, o6, N3);
}
